// LongformerAttention_39865886441713
// MI455X (gfx1250) — matrix operand order repaired and checked by reading, compiled, not run on hardware
//
#include <hip/hip_runtime.h>
#include <hip/hip_bf16.h>

// Longformer attention for MI455X (gfx1250), wave32 + WMMA bf16 + TDM staging.
// B=1, L=4096, D=1024, H=16, HD=64, WINDOW=15 (half-window 7).

#define L_SEQ  4096
#define DMODEL 1024
#define NHEAD  16
#define HDIM   64
#define WHALF  7

typedef __attribute__((ext_vector_type(16))) __bf16 v16bf;
typedef __attribute__((ext_vector_type(8)))  float  v8f;
typedef __attribute__((ext_vector_type(4)))  unsigned int u32x4;
typedef __attribute__((ext_vector_type(8)))  int i32x8;
typedef __attribute__((ext_vector_type(4)))  int i32x4;

union Frag16 {       // one WMMA 16-bit A/B operand: 16 bf16 = 32B = 2x B128
    v16bf v;
    uint4 q[2];
};

// ---------------------------------------------------------------------------
// Tensor Data Mover: 2D tile (tile_k contiguous elems x tile_rows rows of a
// row-major bf16 matrix with row stride row_stride elems) -> LDS.
// D# packing per CDNA5 ISA. Issued per-wave
// (EXEC ignored); tracked by TENSORcnt. 6-arg builtin (clang-23 lane).
// ---------------------------------------------------------------------------
__device__ __forceinline__ void tdm_load_2d(unsigned int lds_byte_addr,
                                            const void* gaddr,
                                            int tile_k, int tile_rows,
                                            int row_stride,
                                            int tensor_d0, int tensor_d1) {
    unsigned long long ga = (unsigned long long)(uintptr_t)gaddr;
    u32x4 g0;
    g0[0] = 1u;                                         // count=1, user mode
    g0[1] = lds_byte_addr;                              // lds_addr [63:32]
    g0[2] = (unsigned int)(ga & 0xFFFFFFFFu);           // global_addr lo
    g0[3] = (unsigned int)((ga >> 32) & 0x01FFFFFFu)    // global_addr hi (57b)
            | (2u << 30);                               // type=2 ("image")
    i32x8 g1;
    g1[0] = (int)(1u << 16);                            // data_size=1 -> 2 bytes
    g1[1] = (int)((unsigned)(tensor_d0 & 0xFFFF) << 16);            // dim0[15:0] @ [63:48]
    g1[2] = (int)(((unsigned)tensor_d0 >> 16) & 0xFFFFu)            // dim0[31:16]
            | (int)((unsigned)(tensor_d1 & 0xFFFF) << 16);          // dim1[15:0]
    g1[3] = (int)(((unsigned)tensor_d1 >> 16) & 0xFFFFu)            // dim1[31:16]
            | (int)((unsigned)tile_k << 16);                        // tile_dim0
    g1[4] = (int)((unsigned)tile_rows & 0xFFFFu);       // tile_dim1 (tile_dim2=0)
    g1[5] = row_stride;                                 // tensor_dim0_stride lo32
    g1[6] = 0;                                          // stride hi16 | dim1_stride lo16
    g1[7] = 0;                                          // dim1_stride hi32
    i32x4 gz4 = {0, 0, 0, 0};                           // groups 2/3: dims/tiles unused
    i32x8 gz8 = {0, 0, 0, 0, 0, 0, 0, 0};               // extra group (clang-23 form)
    __builtin_amdgcn_tensor_load_to_lds(g0, g1, gz4, gz4, gz8, 0);
}

// ---------------------------------------------------------------------------
// f32 -> bf16 convert (elementwise)
// ---------------------------------------------------------------------------
__global__ void lf_convert_bf16(const float* __restrict__ in,
                                __hip_bfloat16* __restrict__ out, int n) {
    int i = blockIdx.x * blockDim.x + threadIdx.x;
    if (i < n) out[i] = __float2bfloat16(in[i]);
}

// ---------------------------------------------------------------------------
// bf16 WMMA GEMM with TDM double-buffered LDS staging:
//   C[M][N] = A[M][K] * W[N][K]^T   (i.e. x @ W.T), M=4096, N=K=1024.
// Block tile 128x64, K-step 64 (two WMMA k-substeps per stage).
// 8 waves (4x2), wave tile 32x32 = 2x2 WMMA tiles.
// ---------------------------------------------------------------------------
__global__ __launch_bounds__(256)
void lf_gemm_bf16_tdm(const __hip_bfloat16* __restrict__ A,
                      const __hip_bfloat16* __restrict__ W,
                      float* __restrict__ C) {
    constexpr int K = DMODEL, N = DMODEL;
    constexpr int KSTEP = 64;
    __shared__ __align__(16) __hip_bfloat16 As[2][128 * KSTEP];  // 2 x 16 KB
    __shared__ __align__(16) __hip_bfloat16 Bs[2][64 * KSTEP];   // 2 x  8 KB

    const int lane = threadIdx.x & 31;
    const int wave = threadIdx.x >> 5;
    const int wm = wave >> 1;                 // 0..3
    const int wn = wave & 1;                  // 0..1
    const int m0 = blockIdx.y * 128;
    const int n0 = blockIdx.x * 64;
    const int hi = lane >> 4;                 // half-wave select
    const int lo = lane & 15;

    const unsigned int asBase = (unsigned int)(uintptr_t)&As[0][0];
    const unsigned int bsBase = (unsigned int)(uintptr_t)&Bs[0][0];

    v8f acc[2][2];
    {
        v8f z = {};
        acc[0][0] = z; acc[0][1] = z; acc[1][0] = z; acc[1][1] = z;
    }

    constexpr int NT = K / KSTEP;             // 16 stages
    if (wave == 0) {                          // prologue: stage 0 in flight
        tdm_load_2d(asBase, A + (size_t)m0 * K, KSTEP, 128, K, K, L_SEQ);
        tdm_load_2d(bsBase, W + (size_t)n0 * K, KSTEP, 64,  K, K, N);
    }

    for (int kt = 0; kt < NT; ++kt) {
        const int cur = kt & 1;
        if (wave == 0) {
            if (kt + 1 < NT) {                // kick next stage, then wait cur
                const int nxt = (kt + 1) & 1;
                const int k0n = (kt + 1) * KSTEP;
                tdm_load_2d(asBase + nxt * (unsigned)sizeof(As[0]),
                            A + (size_t)m0 * K + k0n, KSTEP, 128, K, K, L_SEQ);
                tdm_load_2d(bsBase + nxt * (unsigned)sizeof(Bs[0]),
                            W + (size_t)n0 * K + k0n, KSTEP, 64,  K, K, N);
                __builtin_amdgcn_s_wait_tensorcnt((short)2);
            } else {
                __builtin_amdgcn_s_wait_tensorcnt((short)0);
            }
        }
        __syncthreads();                      // tile visible to all waves

        const __hip_bfloat16* as = As[cur];
        const __hip_bfloat16* bs = Bs[cur];
        #pragma unroll
        for (int ks = 0; ks < 2; ++ks) {      // two 16x16x32 k-substeps
            Frag16 af[2], bg[2];
            #pragma unroll
            for (int t = 0; t < 2; ++t) {
                // A frag (16-bit 16x32 ISA layout): lanes 0-15 K{+0..7,+16..23},
                // lanes 16-31 K{+8..15,+24..31}
                const __hip_bfloat16* ap =
                    as + (wm * 32 + t * 16 + lo) * KSTEP + ks * 32 + 8 * hi;
                af[t].q[0] = *(const uint4*)ap;
                af[t].q[1] = *(const uint4*)(ap + 16);
                // B frag: same K layout as A: lanes 0-15 K{+0..7,+16..23},
                // lanes 16-31 K{+8..15,+24..31}
                const __hip_bfloat16* bp =
                    bs + (wn * 32 + t * 16 + lo) * KSTEP + ks * 32 + 8 * hi;
                bg[t].q[0] = *(const uint4*)bp;
                bg[t].q[1] = *(const uint4*)(bp + 16);
            }
            #pragma unroll
            for (int tm = 0; tm < 2; ++tm)
                #pragma unroll
                for (int tn = 0; tn < 2; ++tn)
                    acc[tm][tn] = __builtin_amdgcn_wmma_f32_16x16x32_bf16(
                        false, af[tm].v, false, bg[tn].v,
                        (short)0, acc[tm][tn], false, false);
        }
        __syncthreads();                      // all reads done before overwrite
    }

    // C/D layout: VGPR r -> row r + 8*hi, lane -> column lo
    #pragma unroll
    for (int tm = 0; tm < 2; ++tm)
        #pragma unroll
        for (int tn = 0; tn < 2; ++tn) {
            float* cp = C + (size_t)(m0 + wm * 32 + tm * 16 + 8 * hi) * N
                          + (n0 + wn * 32 + tn * 16 + lo);
            #pragma unroll
            for (int r = 0; r < 8; ++r)
                cp[(size_t)r * N] = acc[tm][tn][r];
        }
}

// ---------------------------------------------------------------------------
// RoPE in place on Q and K (f32 [L][D], head layout n = h*64+d, pairs (2p,2p+1))
// ---------------------------------------------------------------------------
__global__ void lf_rope(float* __restrict__ Q, float* __restrict__ Kt,
                        const float* __restrict__ fc, const float* __restrict__ fs) {
    int idx = blockIdx.x * blockDim.x + threadIdx.x;   // L*D/2
    if (idx >= L_SEQ * DMODEL / 2) return;
    int l   = idx >> 9;          // /512
    int rem = idx & 511;
    int h   = rem >> 5;          // /32
    int p   = rem & 31;
    float c = fc[l * 32 + p];
    float s = fs[l * 32 + p];
    size_t base = (size_t)l * DMODEL + h * HDIM + 2 * p;
    float a = Q[base], b = Q[base + 1];
    Q[base] = a * c - b * s;  Q[base + 1] = a * s + b * c;
    a = Kt[base]; b = Kt[base + 1];
    Kt[base] = a * c - b * s; Kt[base + 1] = a * s + b * c;
}

// ---------------------------------------------------------------------------
// Sparse attention, rows i >= 1: keys [max(i-7,0), i] plus global key 0.
// One wave per (i,h); lanes parallel over HD (2 elems/lane, wave32);
// online softmax; bf16 output feeds the final WMMA GEMM.
// ---------------------------------------------------------------------------
__global__ __launch_bounds__(256)
void lf_attn_local(const float* __restrict__ Q, const float* __restrict__ Kt,
                   const float* __restrict__ V, __hip_bfloat16* __restrict__ Oa) {
    const int lane = threadIdx.x & 31;
    const int wave = threadIdx.x >> 5;
    int id = blockIdx.x * 8 + wave;
    const int total = (L_SEQ - 1) * NHEAD;
    if (id >= total) return;
    int i = id / NHEAD + 1;
    int h = id % NHEAD;

    const float* qb = Q + (size_t)i * DMODEL + h * HDIM;
    float q0 = qb[lane], q1 = qb[lane + 32];
    float mrun = -3.0e38f, lrun = 0.f, o0 = 0.f, o1 = 0.f;

    auto step = [&](int j) {
        const float* kb = Kt + (size_t)j * DMODEL + h * HDIM;
        float part = q0 * kb[lane] + q1 * kb[lane + 32];
        part += __shfl_xor(part, 16, 32);
        part += __shfl_xor(part, 8, 32);
        part += __shfl_xor(part, 4, 32);
        part += __shfl_xor(part, 2, 32);
        part += __shfl_xor(part, 1, 32);
        float sc = part * 0.125f;                     // 1/sqrt(64)
        float mn = fmaxf(mrun, sc);
        float alpha = __expf(mrun - mn);
        float p = __expf(sc - mn);
        const float* vb = V + (size_t)j * DMODEL + h * HDIM;
        lrun = lrun * alpha + p;
        o0 = o0 * alpha + p * vb[lane];
        o1 = o1 * alpha + p * vb[lane + 32];
        mrun = mn;
    };

    int start = i - WHALF; if (start < 0) start = 0;
    if (start > 0) step(0);                           // global key 0
    for (int j = start; j <= i; ++j) step(j);

    float inv = 1.0f / lrun;
    __hip_bfloat16* ob = Oa + (size_t)i * DMODEL + h * HDIM;
    ob[lane]      = __float2bfloat16(o0 * inv);
    ob[lane + 32] = __float2bfloat16(o1 * inv);
}

// ---------------------------------------------------------------------------
// Row 0 is fully dense (global query): flash-decode style. One block per
// head; 8 waves stride the 4096 keys; merge partials through LDS.
// ---------------------------------------------------------------------------
__global__ __launch_bounds__(256)
void lf_attn_row0(const float* __restrict__ Q, const float* __restrict__ Kt,
                  const float* __restrict__ V, __hip_bfloat16* __restrict__ Oa) {
    __shared__ float sm[8], sl[8], so[8][64];
    const int h = blockIdx.x;
    const int lane = threadIdx.x & 31;
    const int wave = threadIdx.x >> 5;

    const float* qb = Q + h * HDIM;
    float q0 = qb[lane], q1 = qb[lane + 32];
    float mrun = -3.0e38f, lrun = 0.f, o0 = 0.f, o1 = 0.f;

    for (int j = wave; j < L_SEQ; j += 8) {
        const float* kb = Kt + (size_t)j * DMODEL + h * HDIM;
        float part = q0 * kb[lane] + q1 * kb[lane + 32];
        part += __shfl_xor(part, 16, 32);
        part += __shfl_xor(part, 8, 32);
        part += __shfl_xor(part, 4, 32);
        part += __shfl_xor(part, 2, 32);
        part += __shfl_xor(part, 1, 32);
        float sc = part * 0.125f;
        float mn = fmaxf(mrun, sc);
        float alpha = __expf(mrun - mn);
        float p = __expf(sc - mn);
        const float* vb = V + (size_t)j * DMODEL + h * HDIM;
        lrun = lrun * alpha + p;
        o0 = o0 * alpha + p * vb[lane];
        o1 = o1 * alpha + p * vb[lane + 32];
        mrun = mn;
    }
    if (lane == 0) { sm[wave] = mrun; sl[wave] = lrun; }
    so[wave][lane] = o0;
    so[wave][lane + 32] = o1;
    __syncthreads();

    if (wave == 0) {
        float M = -3.0e38f;
        for (int w = 0; w < 8; ++w) M = fmaxf(M, sm[w]);
        float Ltot = 0.f, O0 = 0.f, O1 = 0.f;
        for (int w = 0; w < 8; ++w) {
            float e = __expf(sm[w] - M);
            Ltot += sl[w] * e;
            O0 += so[w][lane] * e;
            O1 += so[w][lane + 32] * e;
        }
        float inv = 1.0f / Ltot;
        __hip_bfloat16* ob = Oa + h * HDIM;
        ob[lane]      = __float2bfloat16(O0 * inv);
        ob[lane + 32] = __float2bfloat16(O1 * inv);
    }
}

// ---------------------------------------------------------------------------
extern "C" void kernel_launch(void* const* d_in, const int* in_sizes, int n_in,
                              void* d_out, int out_size, void* d_ws, size_t ws_size,
                              hipStream_t stream) {
    const float* x  = (const float*)d_in[0];
    const float* Wq = (const float*)d_in[1];
    const float* Wk = (const float*)d_in[2];
    const float* Wv = (const float*)d_in[3];
    const float* Wo = (const float*)d_in[4];
    const float* fc = (const float*)d_in[5];
    const float* fs = (const float*)d_in[6];
    float* out = (float*)d_out;

    // Workspace carve-up (~75.5 MB total)
    char* w = (char*)d_ws;
    __hip_bfloat16* xb  = (__hip_bfloat16*)w; w += (size_t)L_SEQ * DMODEL * 2;
    __hip_bfloat16* wqb = (__hip_bfloat16*)w; w += (size_t)DMODEL * DMODEL * 2;
    __hip_bfloat16* wkb = (__hip_bfloat16*)w; w += (size_t)DMODEL * DMODEL * 2;
    __hip_bfloat16* wvb = (__hip_bfloat16*)w; w += (size_t)DMODEL * DMODEL * 2;
    __hip_bfloat16* wob = (__hip_bfloat16*)w; w += (size_t)DMODEL * DMODEL * 2;
    float* Qr = (float*)w; w += (size_t)L_SEQ * DMODEL * 4;
    float* Kr = (float*)w; w += (size_t)L_SEQ * DMODEL * 4;
    float* Vr = (float*)w; w += (size_t)L_SEQ * DMODEL * 4;
    __hip_bfloat16* attn = (__hip_bfloat16*)w; w += (size_t)L_SEQ * DMODEL * 2;

    const int nXD = L_SEQ * DMODEL;
    const int nDD = DMODEL * DMODEL;

    // 1) f32 -> bf16 conversions for WMMA operands
    lf_convert_bf16<<<(nXD + 255) / 256, 256, 0, stream>>>(x,  xb,  nXD);
    lf_convert_bf16<<<(nDD + 255) / 256, 256, 0, stream>>>(Wq, wqb, nDD);
    lf_convert_bf16<<<(nDD + 255) / 256, 256, 0, stream>>>(Wk, wkb, nDD);
    lf_convert_bf16<<<(nDD + 255) / 256, 256, 0, stream>>>(Wv, wvb, nDD);
    lf_convert_bf16<<<(nDD + 255) / 256, 256, 0, stream>>>(Wo, wob, nDD);

    // 2) Q/K/V projections via WMMA bf16 GEMM with TDM staging
    dim3 gg(DMODEL / 64, L_SEQ / 128);
    lf_gemm_bf16_tdm<<<gg, 256, 0, stream>>>(xb, wqb, Qr);
    lf_gemm_bf16_tdm<<<gg, 256, 0, stream>>>(xb, wkb, Kr);
    lf_gemm_bf16_tdm<<<gg, 256, 0, stream>>>(xb, wvb, Vr);

    // 3) RoPE on Q,K in place
    lf_rope<<<(L_SEQ * DMODEL / 2 + 255) / 256, 256, 0, stream>>>(Qr, Kr, fc, fs);

    // 4) Sparse attention: local windows + global row 0
    lf_attn_local<<<((L_SEQ - 1) * NHEAD + 7) / 8, 256, 0, stream>>>(Qr, Kr, Vr, attn);
    lf_attn_row0<<<NHEAD, 256, 0, stream>>>(Qr, Kr, Vr, attn);

    // 5) Output projection: attn @ Wo.T -> f32 out
    lf_gemm_bf16_tdm<<<gg, 256, 0, stream>>>(attn, wob, out);
}
